// ConstrainedEnhancementModel_4939212390683
// MI455X (gfx1250) — matrix operand order repaired and checked by reading, not compiled, not run on hardware
//
#include <hip/hip_runtime.h>

// ---------------------------------------------------------------------------
// ConstrainedEnhancementModel: 6-layer MLP + spline-blend epilogue, CDNA5 WMMA
// B=512, L=256, F=32, H=4096, HID=512, UP=16
// Layers 1-5: generic 128x128 bf16-WMMA GEMM.
// Layer 6 (512x131072x1024, ~95% of FLOPs): BM=512 (full M) x BN=64 kernel so
// the 512MB w6 streams from HBM exactly once (~33us roofline), activations
// re-read from L2. Blend epilogue fused.
// LDS staging uses packed b64 stores; bf16 via native cvt.
// ---------------------------------------------------------------------------

typedef __attribute__((ext_vector_type(16))) __bf16          v16bf;
typedef __attribute__((ext_vector_type(8)))  float           v8f;
typedef __attribute__((ext_vector_type(8)))  unsigned short  v8us;
typedef __attribute__((ext_vector_type(16))) unsigned short  v16us;

#define BM 128
#define BN 128
#define BK 32
#define LDSS 40   // LDS row stride in bf16 elems (32 + 8 pad) = 80 bytes (16B-aligned)

#define TBM 512   // tall kernel: full M
#define TBN 64

__device__ __forceinline__ unsigned short f2bf(float f) {
  return __builtin_bit_cast(unsigned short, (__bf16)f);  // native cvt, RNE
}

// Pack two floats into one dword of bf16 (lets backend fuse into pk-cvt).
__device__ __forceinline__ unsigned pack2(float a, float b) {
  return (unsigned)f2bf(a) | ((unsigned)f2bf(b) << 16);
}

// Assemble a 16-element bf16 fragment from two 8-element (16-byte) LDS loads.
__device__ __forceinline__ v16bf load_frag(const unsigned short* p, int gap) {
  v8us lo = *(const v8us*)(p);
  v8us hi = *(const v8us*)(p + gap);
  v16us c = __builtin_shufflevector(lo, hi, 0,1,2,3,4,5,6,7,8,9,10,11,12,13,14,15);
  return __builtin_bit_cast(v16bf, c);
}

// Spline-blend epilogue of the reference: n encodes (t = n>>5, f = n&31).
__device__ __forceinline__ float blend_fn(float v, const float* __restrict__ low,
                                          int m, int n) {
  int t   = n >> 5;
  int f   = n & 31;
  int seg = t >> 4;
  int md  = t & 15;
  const float* lb = low + (size_t)m * 8192 + f;   // low[m, :, f], stride 32
  if (md == 0) return lb[seg * 32];               // knot: exact low-res value
  if (t < 4080) {                                  // inside a segment
    int   sl  = seg < 254 ? seg : 254;
    float a   = (float)md * 0.0625f;
    float lin = (1.0f - a) * lb[sl * 32] + a * lb[(sl + 1) * 32];
    return 0.8f * lin + 0.2f * v;
  }
  return v;                                        // tail keeps decoded value
}

// ---------------------------------------------------------------------------
// Generic GEMM: C = act(A[MxK] @ W[KxN] + bias[N]) — layers 1..5
// ---------------------------------------------------------------------------
template<bool RELU>
__global__ __launch_bounds__(256)
void gemm_bf16_wmma(const float* __restrict__ A, const float* __restrict__ W,
                    const float* __restrict__ bias, float* __restrict__ C,
                    int M, int N, int K)
{
  __shared__ unsigned short sA[BM * LDSS];  // [m][k] bf16
  __shared__ unsigned short sB[BN * LDSS];  // [n][k] bf16 (transposed)

  const int tid  = threadIdx.x;
  const int lane = tid & 31;
  const int wave = tid >> 5;       // 0..7
  const int wm   = wave >> 1;      // 0..3 : owns rows [wm*32, wm*32+32)
  const int wn   = wave & 1;       // 0..1 : owns cols [wn*64, wn*64+64)
  const int l15  = lane & 15;
  const int lhi  = lane >> 4;      // 0 or 1

  const int block_m = blockIdx.y * BM;
  const int block_n = blockIdx.x * BN;

  v8f acc[2][4];
  #pragma unroll
  for (int i = 0; i < 2; ++i)
    #pragma unroll
    for (int j = 0; j < 4; ++j) acc[i][j] = (v8f){};

  for (int k0 = 0; k0 < K; k0 += BK) {
    // ---- stage A tile: BM x BK f32 -> bf16 LDS, one packed b64 per float4
    #pragma unroll
    for (int it = 0; it < 4; ++it) {
      int idx = tid + it * 256;
      int row = idx >> 3;               // 0..127
      int c4  = (idx & 7) * 4;          // k offset 0..28
      float4 v = *(const float4*)(A + (size_t)(block_m + row) * K + (k0 + c4));
      uint2 p; p.x = pack2(v.x, v.y); p.y = pack2(v.z, v.w);
      *(uint2*)(sA + row * LDSS + c4) = p;   // 8B-aligned (80B rows)
    }
    // ---- stage B tile: BK x BN f32 -> [n][k] bf16 LDS.
    // Column loads (4x coalesced b32) -> one packed b64 store per thread-item.
    #pragma unroll
    for (int it = 0; it < 4; ++it) {
      int idx = tid + it * 256;
      int n   = idx & 127;              // col within tile
      int kr  = (idx >> 7) * 4;         // 0,4,...,28
      const float* src = W + (size_t)(k0 + kr) * N + (block_n + n);
      float x0 = src[0];
      float x1 = src[(size_t)N];
      float x2 = src[(size_t)N * 2];
      float x3 = src[(size_t)N * 3];
      uint2 p; p.x = pack2(x0, x1); p.y = pack2(x2, x3);
      *(uint2*)(sB + n * LDSS + kr) = p;
    }
    __syncthreads();

    // ---- fragments per documented wave32 VGPR layouts
    v16bf af[2], bf[4];
    #pragma unroll
    for (int i = 0; i < 2; ++i) {
      // A 16x32: lane m = l15; K = {kb..kb+7, kb+16..kb+23}, kb = lhi*8
      const unsigned short* p = sA + (wm * 32 + i * 16 + l15) * LDSS + lhi * 8;
      af[i] = load_frag(p, 16);
    }
    #pragma unroll
    for (int j = 0; j < 4; ++j) {
      // B 32x16: lane n = l15; K = {kb..kb+7, kb+16..kb+23}, kb = lhi*8 (transposed LDS)
      const unsigned short* p = sB + (wn * 64 + j * 16 + l15) * LDSS + lhi * 8;
      bf[j] = load_frag(p, 16);
    }

    #pragma unroll
    for (int i = 0; i < 2; ++i)
      #pragma unroll
      for (int j = 0; j < 4; ++j)
        acc[i][j] = __builtin_amdgcn_wmma_f32_16x16x32_bf16(
            false, af[i], false, bf[j], (short)0, acc[i][j], false, false);

    __syncthreads();
  }

  // ---- epilogue: C/D layout — VGPR r: M = r + lhi*8, N = l15
  #pragma unroll
  for (int i = 0; i < 2; ++i) {
    int m0 = block_m + wm * 32 + i * 16 + lhi * 8;
    #pragma unroll
    for (int j = 0; j < 4; ++j) {
      int n  = block_n + wn * 64 + j * 16 + l15;
      float bn = bias[n];
      #pragma unroll
      for (int r = 0; r < 8; ++r) {
        int m = m0 + r;
        float v = acc[i][j][r] + bn;
        if (RELU) v = fmaxf(v, 0.0f);
        C[(size_t)m * N + n] = v;
      }
    }
  }
}

// ---------------------------------------------------------------------------
// Tall GEMM for layer 6: M=512 covered by one block (w6 streamed once),
// BN=64 strip per block, 512 threads = 16 waves, blend epilogue fused.
// ---------------------------------------------------------------------------
__global__ __launch_bounds__(512)
void gemm_bf16_wmma_tall(const float* __restrict__ A, const float* __restrict__ W,
                         const float* __restrict__ bias, float* __restrict__ C,
                         int N, int K, const float* __restrict__ low)
{
  __shared__ unsigned short sA[TBM * LDSS]; // 512*40*2 = 40 KB
  __shared__ unsigned short sB[TBN * LDSS]; //  64*40*2 =  5 KB

  const int tid  = threadIdx.x;
  const int lane = tid & 31;
  const int wave = tid >> 5;       // 0..15 : owns rows [wave*32, wave*32+32)
  const int l15  = lane & 15;
  const int lhi  = lane >> 4;

  const int block_n = blockIdx.x * TBN;

  v8f acc[2][4];
  #pragma unroll
  for (int i = 0; i < 2; ++i)
    #pragma unroll
    for (int j = 0; j < 4; ++j) acc[i][j] = (v8f){};

  for (int k0 = 0; k0 < K; k0 += BK) {
    // ---- stage A tile: 512 x 32 f32 -> bf16 LDS, packed b64 stores
    #pragma unroll
    for (int it = 0; it < 8; ++it) {
      int idx = tid + it * 512;
      int row = idx >> 3;               // 0..511
      int c4  = (idx & 7) * 4;          // k offset
      float4 v = *(const float4*)(A + (size_t)row * K + (k0 + c4));
      uint2 p; p.x = pack2(v.x, v.y); p.y = pack2(v.z, v.w);
      *(uint2*)(sA + row * LDSS + c4) = p;
    }
    // ---- stage B tile: 32 x 64 f32 -> [n][k] bf16 LDS via column loads
    {
      int n  = tid & 63;                // col within tile
      int kr = (tid >> 6) * 4;          // 0,4,...,28
      const float* src = W + (size_t)(k0 + kr) * N + (block_n + n);
      float x0 = src[0];
      float x1 = src[(size_t)N];
      float x2 = src[(size_t)N * 2];
      float x3 = src[(size_t)N * 3];
      uint2 p; p.x = pack2(x0, x1); p.y = pack2(x2, x3);
      *(uint2*)(sB + n * LDSS + kr) = p;
      // pull the next K-tile of the HBM-resident weight stream ahead
      if (k0 + BK < K) __builtin_prefetch(src + (size_t)BK * N, 0, 1);
    }
    __syncthreads();

    v16bf af[2], bf[4];
    #pragma unroll
    for (int i = 0; i < 2; ++i) {
      const unsigned short* p = sA + (wave * 32 + i * 16 + l15) * LDSS + lhi * 8;
      af[i] = load_frag(p, 16);
    }
    #pragma unroll
    for (int j = 0; j < 4; ++j) {
      const unsigned short* p = sB + (j * 16 + l15) * LDSS + lhi * 8;
      bf[j] = load_frag(p, 16);
    }

    #pragma unroll
    for (int i = 0; i < 2; ++i)
      #pragma unroll
      for (int j = 0; j < 4; ++j)
        acc[i][j] = __builtin_amdgcn_wmma_f32_16x16x32_bf16(
            false, af[i], false, bf[j], (short)0, acc[i][j], false, false);

    __syncthreads();
  }

  // ---- epilogue: bias + spline blend, write to out
  #pragma unroll
  for (int i = 0; i < 2; ++i) {
    int m0 = wave * 32 + i * 16 + lhi * 8;
    #pragma unroll
    for (int j = 0; j < 4; ++j) {
      int n  = block_n + j * 16 + l15;
      float bn = bias[n];
      #pragma unroll
      for (int r = 0; r < 8; ++r) {
        int m = m0 + r;
        float v = blend_fn(acc[i][j][r] + bn, low, m, n);
        C[(size_t)m * N + n] = v;
      }
    }
  }
}

extern "C" void kernel_launch(void* const* d_in, const int* in_sizes, int n_in,
                              void* d_out, int out_size, void* d_ws, size_t ws_size,
                              hipStream_t stream) {
  const float* low = (const float*)d_in[0];   // (512,256,32) == (512,8192)
  const float* w1  = (const float*)d_in[1];   // (8192,1024)
  const float* b1  = (const float*)d_in[2];
  const float* w2  = (const float*)d_in[3];   // (1024,512)
  const float* b2  = (const float*)d_in[4];
  const float* w3  = (const float*)d_in[5];   // (512,256)
  const float* b3  = (const float*)d_in[6];
  const float* w4  = (const float*)d_in[7];   // (256,512)
  const float* b4  = (const float*)d_in[8];
  const float* w5  = (const float*)d_in[9];   // (512,1024)
  const float* b5  = (const float*)d_in[10];
  const float* w6  = (const float*)d_in[11];  // (1024,131072)
  const float* b6  = (const float*)d_in[12];
  float* out = (float*)d_out;                 // (512,4096,32) f32

  const int M = 512;
  float* h1    = (float*)d_ws;                // 512x1024
  float* h2    = h1    + (size_t)M * 1024;    // 512x512
  float* feats = h2    + (size_t)M * 512;     // 512x256
  float* d1    = feats + (size_t)M * 256;     // 512x512
  float* d2    = d1    + (size_t)M * 512;     // 512x1024

  dim3 blk(256);
  auto grid = [](int m, int n) { return dim3((unsigned)(n / BN), (unsigned)(m / BM)); };

  gemm_bf16_wmma<true ><<<grid(M, 1024), blk, 0, stream>>>(low,   w1, b1, h1,    M, 1024, 8192);
  gemm_bf16_wmma<true ><<<grid(M, 512),  blk, 0, stream>>>(h1,    w2, b2, h2,    M, 512,  1024);
  gemm_bf16_wmma<false><<<grid(M, 256),  blk, 0, stream>>>(h2,    w3, b3, feats, M, 256,  512);
  gemm_bf16_wmma<true ><<<grid(M, 512),  blk, 0, stream>>>(feats, w4, b4, d1,    M, 512,  256);
  gemm_bf16_wmma<true ><<<grid(M, 1024), blk, 0, stream>>>(d1,    w5, b5, d2,    M, 1024, 512);

  gemm_bf16_wmma_tall<<<dim3(131072 / TBN), dim3(512), 0, stream>>>(
      d2, w6, b6, out, 131072, 1024, low);

  (void)in_sizes; (void)n_in; (void)out_size; (void)ws_size;
}
